// PhysicsConstrainedFNO_11862699671593
// MI455X (gfx1250) — matrix operand order repaired and checked by reading, not compiled, not run on hardware
//
#include <hip/hip_runtime.h>
#include <hip/hip_bf16.h>
#include <cstdint>

#define Bn 8
#define Cc 64
#define Hh 256
#define Ww 256
#define Kk 16
#define HW (Hh*Ww)
#define NTOT (Bn*Cc*HW)
#define DTc 0.05f
#define EPSc 1e-5f

typedef __attribute__((ext_vector_type(16))) __bf16 v16bf;
typedef __attribute__((ext_vector_type(8)))  float  v8f;
typedef __attribute__((ext_vector_type(4)))  unsigned int uint4v;

union Frag16 { v16bf v; uint4v q[2]; };

__device__ __forceinline__ float gelu_f(float v) {
    return 0.5f * v * (1.0f + erff(v * 0.7071067811865476f));
}

// ---------------------------------------------------------------- tables
__global__ void setup_tables_kernel(float* E_re, float* E_im,
                                    float* F_re, float* F_im,
                                    float* Z_re, float* Z_im,
                                    float* gtab, __bf16* basisT) {
    const float TP = 6.283185307179586f;
    int t = threadIdx.x;  // 0..255
    // E[h] = sum_{ky<16} exp(-2pi i ky h / 256)
    float er = 0.f, ei = 0.f;
    for (int ky = 0; ky < Kk; ky++) {
        float a = -TP * (float)(ky * t) / 256.f;
        er += cosf(a); ei += sinf(a);
    }
    E_re[t] = er; E_im[t] = ei;
    // F[w][kx] = exp(-2pi i kx w / 256) / 256   (ortho 1/sqrt(H*W))
    for (int kx = 0; kx < Kk; kx++) {
        float a = -TP * (float)(kx * t) / 256.f;
        F_re[t * Kk + kx] = cosf(a) * (1.f / 256.f);
        F_im[t * Kk + kx] = sinf(a) * (1.f / 256.f);
    }
    // Z basis: exp(+2pi i ky h / 256) / 16   (ifft over ky, ortho)
    for (int ky = 0; ky < Kk; ky++) {
        float a = TP * (float)(ky * t) / 256.f;
        Z_re[t * Kk + ky] = cosf(a) * (1.f / 16.f);
        Z_im[t * Kk + ky] = sinf(a) * (1.f / 16.f);
    }
    // spectral derivative circulant kernel
    float s = 0.f;
    for (int k = 1; k <= 127; k++) s += (float)k * sinf(TP * (float)(k * t) / 256.f);
    gtab[t] = -2.f * s / 256.f;
    // irfft basis, TRANSPOSED [w][32]: m<16 cos, m>=16 sin (incl. ortho 1/16)
    for (int m = 0; m < 16; m++) {
        float a = TP * (float)(m * t) / 256.f;
        basisT[t * 32 + m]      = (__bf16)(cosf(a) * (1.f / 16.f));
        basisT[t * 32 + 16 + m] = (__bf16)(sinf(a) * (1.f / 16.f));
    }
}

__global__ void convert_weights_kernel(const float* pw, const float* fw, const float* pw1,
                                       __bf16* wpw, __bf16* wflux, __bf16* wpol) {
    int idx = blockIdx.x * 256 + threadIdx.x;   // 24576 total
    if (idx < 16384)       wpw[idx]           = (__bf16)pw[idx];
    else if (idx < 20480)  wflux[idx - 16384] = (__bf16)fw[idx - 16384];
    else if (idx < 24576)  wpol[idx - 20480]  = (__bf16)pw1[idx - 20480];
}

// ---------------------------------------------------------------- lift (K=3, scalar, float4)
__global__ void lift_gelu_kernel(const float* __restrict__ state,
                                 const float* __restrict__ lw,
                                 const float* __restrict__ lb,
                                 float* __restrict__ x) {
    int idx = blockIdx.x * 256 + threadIdx.x;          // NTOT/4
    int p  = (idx & ((HW / 4) - 1)) * 4;
    int co = (idx >> 14) & 63;
    int b  = idx >> 20;
    const float* st = state + (size_t)b * 3 * HW;
    float4 s0 = *(const float4*)(st + p);
    float4 s1 = *(const float4*)(st + HW + p);
    float4 s2 = *(const float4*)(st + 2 * HW + p);
    float w0 = lw[co * 3 + 0], w1 = lw[co * 3 + 1], w2 = lw[co * 3 + 2], bb = lb[co];
    float4 o;
    o.x = gelu_f(bb + w0 * s0.x + w1 * s1.x + w2 * s2.x);
    o.y = gelu_f(bb + w0 * s0.y + w1 * s1.y + w2 * s2.y);
    o.z = gelu_f(bb + w0 * s0.z + w1 * s1.z + w2 * s2.z);
    o.w = gelu_f(bb + w0 * s0.w + w1 * s1.w + w2 * s2.w);
    *(float4*)(x + ((size_t)b * Cc + co) * HW + p) = o;
}

// ---------------------------------------------------------------- spectral pipeline
__global__ void rowreduce_kernel(const float* __restrict__ x,
                                 const float* __restrict__ E_re, const float* __restrict__ E_im,
                                 float* __restrict__ u_re, float* __restrict__ u_im) {
    int idx = blockIdx.x * 256 + threadIdx.x;  // B*C*W/4
    int w4 = (idx & 63) * 4;
    int bc = idx >> 6;
    const float* p = x + (size_t)bc * HW + w4;
    float4 sr = {0.f, 0.f, 0.f, 0.f}, si = {0.f, 0.f, 0.f, 0.f};
    for (int h = 0; h < Hh; h += 8) {
        int hp = h + 24; if (hp > 255) hp = 255;
        __builtin_prefetch(p + (size_t)hp * Ww, 0, 1);
#pragma unroll
        for (int hh = h; hh < h + 8; hh++) {
            float4 v = *(const float4*)(p + (size_t)hh * Ww);
            float er = E_re[hh], ei = E_im[hh];
            sr.x += er * v.x; sr.y += er * v.y; sr.z += er * v.z; sr.w += er * v.w;
            si.x += ei * v.x; si.y += ei * v.y; si.z += ei * v.z; si.w += ei * v.w;
        }
    }
    *(float4*)(u_re + (size_t)bc * Ww + w4) = sr;
    *(float4*)(u_im + (size_t)bc * Ww + w4) = si;
}

__global__ void smode_kernel(const float* __restrict__ u_re, const float* __restrict__ u_im,
                             const float* __restrict__ F_re, const float* __restrict__ F_im,
                             float* __restrict__ S_re, float* __restrict__ S_im) {
    int idx = blockIdx.x * 256 + threadIdx.x;  // B*C*16
    int kx = idx & 15;
    int bc = idx >> 4;
    float sr = 0.f, si = 0.f;
    for (int w = 0; w < Ww; w++) {
        float ur = u_re[bc * Ww + w], ui = u_im[bc * Ww + w];
        float fr = F_re[w * Kk + kx], fi = F_im[w * Kk + kx];
        sr += ur * fr - ui * fi;
        si += ur * fi + ui * fr;
    }
    S_re[idx] = sr; S_im[idx] = si;
}

// blk[b,j,ky,kx] = sum_i S[b,i,ky] * W[i,j,ky,kx]   (einsum label reuse!)
__global__ void blk_kernel(const float* __restrict__ S_re, const float* __restrict__ S_im,
                           const float* __restrict__ wr, const float* __restrict__ wi,
                           float* __restrict__ blk_re, float* __restrict__ blk_im) {
    int idx = blockIdx.x * 256 + threadIdx.x;  // B*64*16*16
    int kx = idx & 15;
    int ky = (idx >> 4) & 15;
    int j  = (idx >> 8) & 63;
    int b  = idx >> 14;
    float sr = 0.f, si = 0.f;
    for (int i = 0; i < Cc; i++) {
        float Sr = S_re[(b * Cc + i) * Kk + ky];
        float Si = S_im[(b * Cc + i) * Kk + ky];
        size_t wo = (((size_t)i * Cc + j) * Kk + ky) * Kk + kx;
        float Wr = wr[wo], Wi = wi[wo];
        sr += Sr * Wr - Si * Wi;
        si += Sr * Wi + Si * Wr;
    }
    blk_re[idx] = sr; blk_im[idx] = si;
}

// coef[b,j,h,0..31]: [ReZ0, 2ReZ1..15, 0, -2ImZ1..15]  (bf16, 4x16B stores)
__global__ void coef_kernel(const float* __restrict__ blk_re, const float* __restrict__ blk_im,
                            const float* __restrict__ Z_re, const float* __restrict__ Z_im,
                            __bf16* __restrict__ coef) {
    int idx = blockIdx.x * 256 + threadIdx.x;  // B*C*H
    int h = idx & (Hh - 1);
    int bj = idx >> 8;
    const float* br = blk_re + (size_t)bj * 256;
    const float* bi = blk_im + (size_t)bj * 256;
    float zr[16], zi[16];
#pragma unroll
    for (int k = 0; k < 16; k++) { zr[k] = 0.f; zi[k] = 0.f; }
    for (int ky = 0; ky < 16; ky++) {
        float er = Z_re[h * Kk + ky], ei = Z_im[h * Kk + ky];
#pragma unroll
        for (int kx = 0; kx < 16; kx++) {
            float r = br[ky * 16 + kx], m = bi[ky * 16 + kx];
            zr[kx] += r * er - m * ei;
            zi[kx] += r * ei + m * er;
        }
    }
    union { __bf16 h16[32]; uint4v q[4]; } pk;
    pk.h16[0]  = (__bf16)zr[0];
    pk.h16[16] = (__bf16)0.f;
#pragma unroll
    for (int kx = 1; kx < 16; kx++) {
        pk.h16[kx]      = (__bf16)(2.f * zr[kx]);
        pk.h16[16 + kx] = (__bf16)(-2.f * zi[kx]);
    }
    uint4v* dst = (uint4v*)(coef + (size_t)idx * 32);
#pragma unroll
    for (int q = 0; q < 4; q++) dst[q] = pk.q[q];
}

// y[b,j,h,w] = sum_{m<32} coef[b,j,h,m] * BasisT[w,m]   -> one WMMA per 16x16 tile
__global__ __launch_bounds__(128) void irfft_wmma_kernel(const __bf16* __restrict__ coef,
                                                         const __bf16* __restrict__ basisT,
                                                         float* __restrict__ out) {
    int lane = threadIdx.x & 31, wave = threadIdx.x >> 5;
    int blk = blockIdx.x;              // B*C*(H/16)
    int bj = blk >> 4;
    int h0 = (blk & 15) * 16;
    const __bf16* crow = coef + (size_t)bj * Hh * 32;
    int m = lane & 15;
    int abase = (lane & 16) ? 8 : 0;   // A frag: runs [abase,abase+8) and [abase+16,abase+24)
    Frag16 fa;
    {
        const uint4v* ap = (const uint4v*)(crow + (h0 + m) * 32 + abase);
        fa.q[0] = ap[0];
        fa.q[1] = ap[2];
    }
    float* ob = out + (size_t)bj * HW;
    int kbase = (lane & 16) ? 8 : 0;   // B frag: runs [kbase,kbase+8) and [kbase+16,kbase+24)
    for (int nt = 0; nt < 4; nt++) {
        int n0 = wave * 64 + nt * 16;
        Frag16 fb;
        const uint4v* bp = (const uint4v*)(basisT + (n0 + (lane & 15)) * 32 + kbase);
        fb.q[0] = bp[0];
        fb.q[1] = bp[2];
        v8f acc = {};
        acc = __builtin_amdgcn_wmma_f32_16x16x32_bf16(false, fa.v, false, fb.v, (short)0, acc, false, false);
#pragma unroll
        for (int r = 0; r < 8; r++) {
            int hh = h0 + r + abase;
            ob[(size_t)hh * Ww + n0 + (lane & 15)] = acc[r];
        }
    }
}

// ---------------------------------------------------------------- 1x1 conv (64->64) via WMMA
// LDS tile stored TRANSPOSED [px][72] (pad 64->72 keeps 16B alignment: 144B stride)
#define LDPAD 72
__global__ __launch_bounds__(256) void pconv_wmma_kernel(const float* __restrict__ in,
                                                         float* __restrict__ out,
                                                         const __bf16* __restrict__ Wb,
                                                         const float* __restrict__ bias,
                                                         int apply_gelu) {
    __shared__ __bf16 lds[32 * LDPAD];
    int tile = blockIdx.x;                       // B * (HW/32)
    int b = tile / (HW / 32);
    int p0 = (tile % (HW / 32)) * 32;
    const float* inb = in + (size_t)b * Cc * HW;
#pragma unroll
    for (int r = 0; r < 8; r++) {
        int idx = threadIdx.x + 256 * r;         // 0..2047
        int ci = idx >> 5, px = idx & 31;
        lds[px * LDPAD + ci] = (__bf16)inb[(size_t)ci * HW + p0 + px];
    }
    __syncthreads();
    int lane = threadIdx.x & 31, wave = threadIdx.x >> 5;
    int co0 = (wave & 3) * 16;
    int n0  = (wave >> 2) * 16;
    int abase = (lane & 16) ? 8 : 0;
    int kbase = (lane & 16) ? 8 : 0;
    int mrow = co0 + (lane & 15);
    Frag16 a0, a1, b0, b1;
    {
        const uint4v* wp = (const uint4v*)(Wb + mrow * 64 + abase);
        a0.q[0] = wp[0]; a0.q[1] = wp[2];             // K runs abase.., abase+16..
        const uint4v* wp2 = (const uint4v*)(Wb + mrow * 64 + 32 + abase);
        a1.q[0] = wp2[0]; a1.q[1] = wp2[2];
        const uint4v* bp = (const uint4v*)(&lds[(n0 + (lane & 15)) * LDPAD + kbase]);
        b0.q[0] = bp[0]; b0.q[1] = bp[2];             // K runs kbase.., kbase+16..
        const uint4v* bp2 = (const uint4v*)(&lds[(n0 + (lane & 15)) * LDPAD + 32 + kbase]);
        b1.q[0] = bp2[0]; b1.q[1] = bp2[2];
    }
    v8f acc = {};
    acc = __builtin_amdgcn_wmma_f32_16x16x32_bf16(false, a0.v, false, b0.v, (short)0, acc, false, false);
    acc = __builtin_amdgcn_wmma_f32_16x16x32_bf16(false, a1.v, false, b1.v, (short)0, acc, false, false);
    float* outb = out + (size_t)b * Cc * HW;
#pragma unroll
    for (int r = 0; r < 8; r++) {
        int mm = co0 + r + abase;
        float v = acc[r] + bias[mm];
        if (apply_gelu) v = gelu_f(v);
        outb[(size_t)mm * HW + p0 + n0 + (lane & 15)] = v;
    }
}

// ---------------------------------------------------------------- instance norm
__global__ __launch_bounds__(256) void stats_kernel(const float* __restrict__ in,
                                                    float* __restrict__ stats) {
    int bc = blockIdx.x;                 // B*C
    const float* p = in + (size_t)bc * HW;
    float s = 0.f, ss = 0.f;
    for (int i = threadIdx.x * 4; i < HW; i += 1024) {
        float4 v = *(const float4*)(p + i);
        s  += v.x + v.y + v.z + v.w;
        ss += v.x * v.x + v.y * v.y + v.z * v.z + v.w * v.w;
    }
    __shared__ float sh[512];
    sh[threadIdx.x] = s; sh[256 + threadIdx.x] = ss;
    __syncthreads();
    for (int o = 128; o > 0; o >>= 1) {
        if (threadIdx.x < o) {
            sh[threadIdx.x] += sh[threadIdx.x + o];
            sh[256 + threadIdx.x] += sh[256 + threadIdx.x + o];
        }
        __syncthreads();
    }
    if (threadIdx.x == 0) {
        float mu = sh[0] * (1.f / HW);
        float var = sh[256] * (1.f / HW) - mu * mu;
        stats[bc * 2] = mu;
        stats[bc * 2 + 1] = rsqrtf(var + EPSc);
    }
}

__global__ void norm_gelu_res_kernel(const float* __restrict__ y,
                                     const float* __restrict__ stats,
                                     float* __restrict__ x) {
    int idx = blockIdx.x * 256 + threadIdx.x;  // NTOT/4
    int bc = idx >> 14;
    float mu = stats[bc * 2], rs = stats[bc * 2 + 1];
    size_t e = (size_t)idx * 4;
    float4 v = *(const float4*)(y + e);
    float4 r = *(const float4*)(x + e);
    float4 o;
    o.x = gelu_f((v.x - mu) * rs) + r.x;
    o.y = gelu_f((v.y - mu) * rs) + r.y;
    o.z = gelu_f((v.z - mu) * rs) + r.z;
    o.w = gelu_f((v.w - mu) * rs) + r.w;
    *(float4*)(x + e) = o;
}

// ---------------------------------------------------------------- 64->2 head (float4 over pixels)
__global__ void head2_kernel(const float* __restrict__ in,
                             const float* __restrict__ w2, const float* __restrict__ b2,
                             float* __restrict__ out) {
    int idx = blockIdx.x * 256 + threadIdx.x;  // B*HW/4
    int b = idx >> 14;
    int p = (idx & ((HW / 4) - 1)) * 4;
    const float* ib = in + (size_t)b * Cc * HW + p;
    float4 s0 = {b2[0], b2[0], b2[0], b2[0]};
    float4 s1 = {b2[1], b2[1], b2[1], b2[1]};
    for (int ci = 0; ci < Cc; ci++) {
        float4 v = *(const float4*)(ib + (size_t)ci * HW);
        float a = w2[ci], c = w2[64 + ci];
        s0.x += a * v.x; s0.y += a * v.y; s0.z += a * v.z; s0.w += a * v.w;
        s1.x += c * v.x; s1.y += c * v.y; s1.z += c * v.z; s1.w += c * v.w;
    }
    *(float4*)(out + (size_t)b * 2 * HW + p)      = s0;
    *(float4*)(out + (size_t)b * 2 * HW + HW + p) = s1;
}

// div = g *_h Jx + g *_w Jy  (spectral derivative as circulant convolution)
__global__ __launch_bounds__(256) void divergence_kernel(const float* __restrict__ J,
                                                         const float* __restrict__ gtab,
                                                         float* __restrict__ dv) {
    int b = blockIdx.x >> 8;
    int h = blockIdx.x & 255;
    int w = threadIdx.x;
    __shared__ float jy[256];
    __shared__ float gs[256];
    const float* Jx = J + (size_t)b * 2 * HW;
    const float* Jy = Jx + HW;
    jy[w] = Jy[h * Ww + w];
    gs[w] = gtab[w];
    __syncthreads();
    float s = 0.f;
    for (int hp = 0; hp < Hh; hp += 8) {
        int pf = hp + 24; if (pf > 255) pf = 255;
        __builtin_prefetch(Jx + (size_t)pf * Ww + w, 0, 1);
#pragma unroll
        for (int hh = hp; hh < hp + 8; hh++)
            s += gs[(h - hh) & 255] * Jx[(size_t)hh * Ww + w];
    }
    for (int wp = 0; wp < Ww; wp++) s += gs[(w - wp) & 255] * jy[wp];
    dv[(size_t)b * HW + h * Ww + w] = s;
}

__global__ void finalize_kernel(const float* __restrict__ state,
                                const float* __restrict__ dv,
                                const float* __restrict__ P,
                                float* __restrict__ out) {
    int idx = blockIdx.x * 256 + threadIdx.x;  // B*HW/4
    int b = idx >> 14;
    int p = (idx & ((HW / 4) - 1)) * 4;
    const float* st = state + (size_t)b * 3 * HW;
    float4 dvv = *(const float4*)(dv + (size_t)b * HW + p);
    float4 dp0 = *(const float4*)(P + (size_t)b * 2 * HW + p);
    float4 dp1 = *(const float4*)(P + (size_t)b * 2 * HW + HW + p);
    float4 st0 = *(const float4*)(st + p);
    float4 st1 = *(const float4*)(st + HW + p);
    float4 st2 = *(const float4*)(st + 2 * HW + p);
    float4 dtr = {-dvv.x, -dvv.y, -dvv.z, -dvv.w};
    float4 rho;
    {
        float zz[4] = {st0.x + DTc * dtr.x, st0.y + DTc * dtr.y,
                       st0.z + DTc * dtr.z, st0.w + DTc * dtr.w};
        float rr[4];
#pragma unroll
        for (int k = 0; k < 4; k++) {
            float z = 10.f * zz[k];
            rr[k] = (fmaxf(z, 0.f) + log1pf(expf(-fabsf(z)))) * 0.1f;  // logaddexp(0,z)/10
        }
        rho.x = rr[0]; rho.y = rr[1]; rho.z = rr[2]; rho.w = rr[3];
    }
    float4 p0n = {st1.x + DTc * dp0.x, st1.y + DTc * dp0.y, st1.z + DTc * dp0.z, st1.w + DTc * dp0.w};
    float4 p1n = {st2.x + DTc * dp1.x, st2.y + DTc * dp1.y, st2.z + DTc * dp1.z, st2.w + DTc * dp1.w};
    float* o1 = out + (size_t)b * 3 * HW;
    float* o2 = out + (size_t)Bn * 3 * HW + (size_t)b * 3 * HW;
    *(float4*)(o1 + p)          = rho;
    *(float4*)(o1 + HW + p)     = p0n;
    *(float4*)(o1 + 2 * HW + p) = p1n;
    *(float4*)(o2 + p)          = dtr;
    *(float4*)(o2 + HW + p)     = dp0;
    *(float4*)(o2 + 2 * HW + p) = dp1;
}

// ----------------------------------------------------------------
extern "C" void kernel_launch(void* const* d_in, const int* in_sizes, int n_in,
                              void* d_out, int out_size, void* d_ws, size_t ws_size,
                              hipStream_t stream) {
    const float* state   = (const float*)d_in[0];
    const float* lift_w  = (const float*)d_in[1];
    const float* lift_b  = (const float*)d_in[2];
    const float* sw_real = (const float*)d_in[3];
    const float* sw_imag = (const float*)d_in[4];
    const float* pw_w    = (const float*)d_in[5];
    const float* pw_b    = (const float*)d_in[6];
    const float* flux_w1 = (const float*)d_in[7];
    const float* flux_b1 = (const float*)d_in[8];
    const float* flux_w2 = (const float*)d_in[9];
    const float* flux_b2 = (const float*)d_in[10];
    const float* pol_w1  = (const float*)d_in[11];
    const float* pol_b1  = (const float*)d_in[12];
    const float* pol_w2  = (const float*)d_in[13];
    const float* pol_b2  = (const float*)d_in[14];
    float* out = (float*)d_out;

    char* wsb = (char*)d_ws;
    auto take = [&](size_t bytes) -> char* {
        char* p = wsb;
        wsb += (bytes + 255) & ~(size_t)255;
        return p;
    };
    float* xbuf   = (float*)take((size_t)NTOT * 4);
    float* ybuf   = (float*)take((size_t)NTOT * 4);
    float* u_re   = (float*)take((size_t)Bn * Cc * Ww * 4);
    float* u_im   = (float*)take((size_t)Bn * Cc * Ww * 4);
    float* S_re   = (float*)take((size_t)Bn * Cc * Kk * 4);
    float* S_im   = (float*)take((size_t)Bn * Cc * Kk * 4);
    float* blk_re = (float*)take((size_t)Bn * Cc * Kk * Kk * 4);
    float* blk_im = (float*)take((size_t)Bn * Cc * Kk * Kk * 4);
    float* stats  = (float*)take((size_t)Bn * Cc * 2 * 4);
    float* E_re   = (float*)take(256 * 4);
    float* E_im   = (float*)take(256 * 4);
    float* F_re   = (float*)take(256 * Kk * 4);
    float* F_im   = (float*)take(256 * Kk * 4);
    float* Z_re   = (float*)take(256 * Kk * 4);
    float* Z_im   = (float*)take(256 * Kk * 4);
    float* gtab   = (float*)take(256 * 4);
    float* Jbuf   = (float*)take((size_t)Bn * 2 * HW * 4);
    float* dvbuf  = (float*)take((size_t)Bn * HW * 4);
    float* Pbuf   = (float*)take((size_t)Bn * 2 * HW * 4);
    __bf16* coef   = (__bf16*)take((size_t)Bn * Cc * Hh * 32 * 2);
    __bf16* basisT = (__bf16*)take(256 * 32 * 2);
    __bf16* wpw    = (__bf16*)take(4 * 64 * 64 * 2);
    __bf16* wflux  = (__bf16*)take(64 * 64 * 2);
    __bf16* wpol   = (__bf16*)take(64 * 64 * 2);

    setup_tables_kernel<<<1, 256, 0, stream>>>(E_re, E_im, F_re, F_im, Z_re, Z_im, gtab, basisT);
    convert_weights_kernel<<<96, 256, 0, stream>>>(pw_w, flux_w1, pol_w1, wpw, wflux, wpol);

    lift_gelu_kernel<<<NTOT / 4 / 256, 256, 0, stream>>>(state, lift_w, lift_b, xbuf);

    for (int l = 0; l < 4; l++) {
        const float* wr = sw_real + (size_t)l * Cc * Cc * Kk * Kk;
        const float* wi = sw_imag + (size_t)l * Cc * Cc * Kk * Kk;
        rowreduce_kernel<<<(Bn * Cc * Ww / 4) / 256, 256, 0, stream>>>(xbuf, E_re, E_im, u_re, u_im);
        smode_kernel<<<(Bn * Cc * Kk) / 256, 256, 0, stream>>>(u_re, u_im, F_re, F_im, S_re, S_im);
        blk_kernel<<<(Bn * Cc * Kk * Kk) / 256, 256, 0, stream>>>(S_re, S_im, wr, wi, blk_re, blk_im);
        coef_kernel<<<(Bn * Cc * Hh) / 256, 256, 0, stream>>>(blk_re, blk_im, Z_re, Z_im, coef);
        irfft_wmma_kernel<<<Bn * Cc * (Hh / 16), 128, 0, stream>>>(coef, basisT, ybuf);
        pconv_wmma_kernel<<<Bn * (HW / 32), 256, 0, stream>>>(ybuf, ybuf, wpw + (size_t)l * 4096,
                                                              pw_b + (size_t)l * 64, 0);
        stats_kernel<<<Bn * Cc, 256, 0, stream>>>(ybuf, stats);
        norm_gelu_res_kernel<<<NTOT / 4 / 256, 256, 0, stream>>>(ybuf, stats, xbuf);
    }

    // flux head -> J -> divergence
    pconv_wmma_kernel<<<Bn * (HW / 32), 256, 0, stream>>>(xbuf, ybuf, wflux, flux_b1, 1);
    head2_kernel<<<(Bn * HW / 4) / 256, 256, 0, stream>>>(ybuf, flux_w2, flux_b2, Jbuf);
    divergence_kernel<<<Bn * Hh, 256, 0, stream>>>(Jbuf, gtab, dvbuf);

    // polarization head
    pconv_wmma_kernel<<<Bn * (HW / 32), 256, 0, stream>>>(xbuf, ybuf, wpol, pol_b1, 1);
    head2_kernel<<<(Bn * HW / 4) / 256, 256, 0, stream>>>(ybuf, pol_w2, pol_b2, Pbuf);

    finalize_kernel<<<(Bn * HW / 4) / 256, 256, 0, stream>>>(state, dvbuf, Pbuf, out);
}
